// GNNClassifier_88648124990545
// MI455X (gfx1250) — matrix operand order repaired and checked by reading, not compiled, not run on hardware
//
#include <hip/hip_runtime.h>

typedef __attribute__((ext_vector_type(16))) _Float16 v16h;
typedef __attribute__((ext_vector_type(8)))  _Float16 v8h;
typedef __attribute__((ext_vector_type(8)))  float    v8f;

#define DD 64

__device__ __forceinline__ void atomAddF(float* p, float v) {
    __hip_atomic_fetch_add(p, v, __ATOMIC_RELAXED, __HIP_MEMORY_SCOPE_AGENT);
}

// h[i,:] = emb[x[i],:]  (float4 per thread)
__global__ __launch_bounds__(256) void gnn_embed(const int* __restrict__ x,
                                                 const float* __restrict__ emb,
                                                 float* __restrict__ h, int n_nodes) {
    int t = blockIdx.x * blockDim.x + threadIdx.x;
    int nd = t >> 4, c = (t & 15) << 2;
    if (nd >= n_nodes) return;
    int tok = x[nd];
    *(float4*)(h + (size_t)nd * DD + c) = *(const float4*)(emb + (size_t)tok * DD + c);
}

// cnt[dst[e]] += 1
__global__ __launch_bounds__(256) void gnn_degree(const int* __restrict__ dst,
                                                  float* __restrict__ cnt, int n_edges) {
    int e = blockIdx.x * blockDim.x + threadIdx.x;
    if (e < n_edges) atomAddF(cnt + dst[e], 1.0f);
}

// agg[dst[e],:] += h[src[e],:]  (16 lanes per edge, float4 each)
__global__ __launch_bounds__(256) void gnn_scatter(const float* __restrict__ h,
                                                   const int* __restrict__ src,
                                                   const int* __restrict__ dst,
                                                   float* __restrict__ agg, int n_edges) {
    int t = blockIdx.x * blockDim.x + threadIdx.x;
    int e = t >> 4, c = (t & 15) << 2;
    if (e >= n_edges) return;
    int s = src[e], d = dst[e];
    float4 v = *(const float4*)(h + (size_t)s * DD + c);
    float* p = agg + (size_t)d * DD + c;
    atomAddF(p + 0, v.x); atomAddF(p + 1, v.y);
    atomAddF(p + 2, v.z); atomAddF(p + 3, v.w);
}

// Fused SAGE update: out = relu( (agg/max(cnt,1)) @ Wl^T + bl + h @ Wr^T )
// 8 waves per block, one 16-node tile per wave; WMMA f32<-f16 16x16x32.
// D is staged through LDS so global stores are coalesced float4.
// `out` may alias `agg` (each node's agg row is consumed into LDS before any write).
__global__ __launch_bounds__(256) void gnn_sage_wmma(const float* __restrict__ h,
                                                     const float* __restrict__ agg,
                                                     const float* __restrict__ cnt,
                                                     const float* __restrict__ Wl,
                                                     const float* __restrict__ bl,
                                                     const float* __restrict__ Wr,
                                                     float* __restrict__ out, int n_nodes) {
    __shared__ __align__(32) _Float16 sWl[DD * DD];
    __shared__ __align__(32) _Float16 sWr[DD * DD];
    __shared__ __align__(32) _Float16 sA[8][16 * DD];   // mean-aggregated tile (f16)
    __shared__ __align__(32) _Float16 sH[8][16 * DD];   // root tile (f16)
    __shared__ __align__(16) float    sD[8][16 * DD];   // output staging (f32)
    const int tid  = threadIdx.x;
    const int wave = tid >> 5;
    const int lane = tid & 31;
    const int node0 = blockIdx.x * 128 + wave * 16;
    const bool full = (node0 + 16 <= n_nodes);

    for (int i = tid; i < DD * DD; i += 256) {
        sWl[i] = (_Float16)Wl[i];
        sWr[i] = (_Float16)Wr[i];
    }
    if (full) {
        for (int i = lane; i < 16 * DD; i += 32) {
            int m = i >> 6, k = i & 63;
            int node = node0 + m;
            float hv = h[(size_t)node * DD + k];
            float c  = cnt[node]; c = c > 1.0f ? c : 1.0f;
            float av = agg[(size_t)node * DD + k] * __builtin_amdgcn_rcpf(c);
            sH[wave][i] = (_Float16)hv;
            sA[wave][i] = (_Float16)av;
        }
    } else {
        for (int i = lane; i < 16 * DD; i += 32) {
            int m = i >> 6, k = i & 63;
            int node = node0 + m;
            float hv = 0.0f, av = 0.0f;
            if (node < n_nodes) {
                hv = h[(size_t)node * DD + k];
                float c = cnt[node]; c = c > 1.0f ? c : 1.0f;
                av = agg[(size_t)node * DD + k] * __builtin_amdgcn_rcpf(c);
            }
            sH[wave][i] = (_Float16)hv;
            sA[wave][i] = (_Float16)av;
        }
    }
    __syncthreads();

    const int m    = lane & 15;            // A row / D column index base
    const int hiA  = (lane >> 4) << 4;     // A K-offset: 0 or 16
    const int hiB  = (lane >> 4) << 4;     // B K-offset: 0 or 16
    const int mrow = (lane >> 4) << 3;     // D row offset: 0 or 8

    for (int jt = 0; jt < 4; ++jt) {
        const int j = jt * 16 + m;         // output feature column for this lane
        float bias = bl[j];
        v8f c;
#pragma unroll
        for (int r = 0; r < 8; ++r) c[r] = bias;

#pragma unroll
        for (int kc = 0; kc < 2; ++kc) {
            const int kb = kc * 32;
            // A fragments: 16 contiguous halves per lane (matches B), two runs of 8
            v8h aAlo = *(const v8h*)&sA[wave][m * DD + kb + hiA];
            v8h aAhi = *(const v8h*)&sA[wave][m * DD + kb + 8 + hiA];
            v16h aA = __builtin_shufflevector(aAlo, aAhi,
                       0,1,2,3,4,5,6,7,8,9,10,11,12,13,14,15);
            v8h aHlo = *(const v8h*)&sH[wave][m * DD + kb + hiA];
            v8h aHhi = *(const v8h*)&sH[wave][m * DD + kb + 8 + hiA];
            v16h aH = __builtin_shufflevector(aHlo, aHhi,
                       0,1,2,3,4,5,6,7,8,9,10,11,12,13,14,15);
            // B fragments (B[k,n] = W[n,k]): 16 contiguous halves per lane
            v16h bL = *(const v16h*)&sWl[j * DD + kb + hiB];
            v16h bR = *(const v16h*)&sWr[j * DD + kb + hiB];

            c = __builtin_amdgcn_wmma_f32_16x16x32_f16(false, aA, false, bL,
                                                       (short)0, c, false, false);
            c = __builtin_amdgcn_wmma_f32_16x16x32_f16(false, aH, false, bR,
                                                       (short)0, c, false, false);
        }
        // stage ReLU'd D tile into LDS (conflict-free: lanes hit distinct columns)
#pragma unroll
        for (int r = 0; r < 8; ++r) {
            float v = c[r];
            sD[wave][(mrow + r) * DD + j] = v > 0.0f ? v : 0.0f;
        }
    }

    // coalesced float4 write-back of the 16x64 tile
    if (full) {
        for (int i = lane; i < 256; i += 32) {
            int row = i >> 4, c4 = (i & 15) << 2;
            *(float4*)(out + (size_t)(node0 + row) * DD + c4) =
                *(const float4*)&sD[wave][row * DD + c4];
        }
    } else {
        for (int i = lane; i < 256; i += 32) {
            int row = i >> 4, c4 = (i & 15) << 2;
            int node = node0 + row;
            if (node < n_nodes)
                *(float4*)(out + (size_t)node * DD + c4) =
                    *(const float4*)&sD[wave][row * DD + c4];
        }
    }
}

// pooled[batch[i],:] += h[i,:] ; gcnt[batch[i]] += 1
__global__ __launch_bounds__(256) void gnn_pool(const float* __restrict__ h,
                                                const int* __restrict__ batch,
                                                float* __restrict__ pooled,
                                                float* __restrict__ gcnt, int n_nodes) {
    int t = blockIdx.x * blockDim.x + threadIdx.x;
    int nd = t >> 4, c = (t & 15) << 2;
    if (nd >= n_nodes) return;
    int g = batch[nd];
    float4 v = *(const float4*)(h + (size_t)nd * DD + c);
    float* p = pooled + (size_t)g * DD + c;
    atomAddF(p + 0, v.x); atomAddF(p + 1, v.y);
    atomAddF(p + 2, v.z); atomAddF(p + 3, v.w);
    if (c == 0) atomAddF(gcnt + g, 1.0f);
}

// out[g,:] = (pooled[g,:]/max(gcnt,1)) @ Wout^T + bout   (one wave per graph)
__global__ __launch_bounds__(32) void gnn_readout(const float* __restrict__ pooled,
                                                  const float* __restrict__ gcnt,
                                                  const float* __restrict__ Wout,
                                                  const float* __restrict__ bout,
                                                  float* __restrict__ out) {
    int g = blockIdx.x;
    int lane = threadIdx.x;
    float c = gcnt[g]; c = c > 1.0f ? c : 1.0f;
    float v0 = pooled[(size_t)g * DD + lane] / c;
    float v1 = pooled[(size_t)g * DD + lane + 32] / c;
    float p0 = v0 * Wout[lane]      + v1 * Wout[lane + 32];
    float p1 = v0 * Wout[64 + lane] + v1 * Wout[96 + lane];
#pragma unroll
    for (int off = 16; off > 0; off >>= 1) {
        p0 += __shfl_down(p0, off, 32);
        p1 += __shfl_down(p1, off, 32);
    }
    if (lane == 0) {
        out[(size_t)g * 2 + 0] = p0 + bout[0];
        out[(size_t)g * 2 + 1] = p1 + bout[1];
    }
}

extern "C" void kernel_launch(void* const* d_in, const int* in_sizes, int n_in,
                              void* d_out, int out_size, void* d_ws, size_t ws_size,
                              hipStream_t stream) {
    const int*   x     = (const int*)d_in[0];
    const int*   ei    = (const int*)d_in[1];
    const int*   batch = (const int*)d_in[2];
    const float* emb   = (const float*)d_in[3];
    const float* W1l   = (const float*)d_in[4];
    const float* b1l   = (const float*)d_in[5];
    const float* W1r   = (const float*)d_in[6];
    const float* W2l   = (const float*)d_in[7];
    const float* b2l   = (const float*)d_in[8];
    const float* W2r   = (const float*)d_in[9];
    const float* Wout  = (const float*)d_in[10];
    const float* bout  = (const float*)d_in[11];
    float* out = (float*)d_out;

    const int n_nodes  = in_sizes[0];
    const int n_edges  = in_sizes[1] / 2;
    const int n_graphs = out_size / 2;
    const int* src = ei;
    const int* dst = ei + n_edges;

    // Workspace layout: two big ping-pong node buffers + small scratch
    const size_t S = (size_t)n_nodes * DD * sizeof(float);
    char* ws = (char*)d_ws;
    float* bufA   = (float*)(ws);              // h0, later agg2/h2 (aliased in sage)
    float* bufB   = (float*)(ws + S);          // agg1/h1 (aliased in sage)
    size_t off = 2 * S;
    float* cnt    = (float*)(ws + off);  off += (((size_t)n_nodes * 4 + 255) / 256) * 256;
    float* pooled = (float*)(ws + off);  off += (size_t)n_graphs * DD * sizeof(float);
    float* gcnt   = (float*)(ws + off);

    const int thr = 256;
    dim3 gEmbed(((unsigned)n_nodes * 16 + thr - 1) / thr);
    dim3 gEdge (((unsigned)n_edges + thr - 1) / thr);
    dim3 gScat (((unsigned)n_edges * 16 + thr - 1) / thr);
    dim3 gSage (((unsigned)n_nodes + 127) / 128);
    dim3 gPool (((unsigned)n_nodes * 16 + thr - 1) / thr);

    // degree (shared by both layers) + embedding
    hipMemsetAsync(cnt, 0, (size_t)n_nodes * sizeof(float), stream);
    gnn_embed<<<gEmbed, thr, 0, stream>>>(x, emb, bufA, n_nodes);
    gnn_degree<<<gEdge, thr, 0, stream>>>(dst, cnt, n_edges);

    // Layer 1: agg in bufB, h1 overwrites bufB (safe block-local aliasing)
    hipMemsetAsync(bufB, 0, S, stream);
    gnn_scatter<<<gScat, thr, 0, stream>>>(bufA, src, dst, bufB, n_edges);
    gnn_sage_wmma<<<gSage, thr, 0, stream>>>(bufA, bufB, cnt, W1l, b1l, W1r, bufB, n_nodes);

    // Layer 2: agg in bufA (h0 dead), h2 overwrites bufA
    hipMemsetAsync(bufA, 0, S, stream);
    gnn_scatter<<<gScat, thr, 0, stream>>>(bufB, src, dst, bufA, n_edges);
    gnn_sage_wmma<<<gSage, thr, 0, stream>>>(bufB, bufA, cnt, W2l, b2l, W2r, bufA, n_nodes);

    // Global mean pool + readout
    hipMemsetAsync(pooled, 0, (size_t)n_graphs * DD * sizeof(float), stream);
    hipMemsetAsync(gcnt, 0, (size_t)n_graphs * sizeof(float), stream);
    gnn_pool<<<gPool, thr, 0, stream>>>(bufA, batch, pooled, gcnt, n_nodes);
    gnn_readout<<<dim3((unsigned)n_graphs), 32, 0, stream>>>(pooled, gcnt, Wout, bout, out);
}
